// AttnBlock_51204600103553
// MI455X (gfx1250) — compile-verified, matrix operand order checked by reading, not run on hardware
//
#include <hip/hip_runtime.h>
#include <hip/hip_bf16.h>

#define NB 2
#define NC 512
#define NHW 4096
#define NG 32
#define CG 16
#define GSZ (CG * NHW)   // 65536 elements per group

typedef __attribute__((ext_vector_type(16))) __bf16 bf16x16;
typedef __attribute__((ext_vector_type(8)))  float  f32x8;
typedef unsigned short u16b;
typedef unsigned int   u32b;

__device__ __forceinline__ u16b f2bf(float f) {
    u32b u = __float_as_uint(f);
    u32b r = u + 0x7FFFu + ((u >> 16) & 1u);   // round-to-nearest-even
    return (u16b)(r >> 16);
}

// CDNA5 async copy: one 16-byte chunk global -> LDS per lane (ASYNCcnt).
// Generic LDS pointers carry the LDS byte offset in addr[31:0] (ISA 10.2).
__device__ __forceinline__ void async_copy16(u16b* lds, const u16b* g) {
    asm volatile("global_load_async_to_lds_b128 %0, %1, off"
                 :: "v"((u32b)(size_t)lds), "v"(g) : "memory");
}
__device__ __forceinline__ void async_wait0() {
    asm volatile("s_wait_asynccnt 0x0" ::: "memory");
}

// 16x32 bf16 WMMA fragment (A layout; also B when tile is stored [n][k]
// row-major). Lane l: row m=l&15, element e: k = 8*(l>>4)+(e&7)+16*(e>>3)
// -> two contiguous 16B runs -> two b128 loads.
__device__ __forceinline__ bf16x16 load_frag(const u16b* base, int stride, int lane) {
    const u16b* p = base + (lane & 15) * stride + ((lane >> 4) << 3);
    union { bf16x16 v; uint4 q[2]; } u;
    u.q[0] = *(const uint4*)(p);
    u.q[1] = *(const uint4*)(p + 16);
    return u.v;
}

__device__ __forceinline__ f32x8 wmma_bf16(bf16x16 a, bf16x16 b, f32x8 c) {
    return __builtin_amdgcn_wmma_f32_16x16x32_bf16(false, a, false, b, (short)0, c, false, false);
}

// ---------------------------------------------------------------- GroupNorm
__global__ void __launch_bounds__(256) k_gn_stats(const float* __restrict__ x,
                                                  float* __restrict__ mr) {
    __shared__ float s0[256], s1[256];
    const int g = blockIdx.x;
    const float* p = x + (size_t)g * GSZ;
    float s = 0.f, ss = 0.f;
    for (int i = threadIdx.x; i < GSZ; i += 256) { float v = p[i]; s += v; ss += v * v; }
    s0[threadIdx.x] = s; s1[threadIdx.x] = ss;
    __syncthreads();
    for (int o = 128; o > 0; o >>= 1) {
        if (threadIdx.x < o) { s0[threadIdx.x] += s0[threadIdx.x + o]; s1[threadIdx.x] += s1[threadIdx.x + o]; }
        __syncthreads();
    }
    if (threadIdx.x == 0) {
        float m = s0[0] * (1.f / GSZ);
        float var = s1[0] * (1.f / GSZ) - m * m;
        mr[2 * g] = m;
        mr[2 * g + 1] = rsqrtf(var + 1e-6f);
    }
}

// GroupNorm apply + transpose: xnT[b][n][c] = gn(x[b][c][n]) in bf16.
// 64n x 64c tile through LDS so global reads AND writes stay coalesced.
__global__ void __launch_bounds__(256) k_gn_apply_t(const float* __restrict__ x,
                                                    const float* __restrict__ gw,
                                                    const float* __restrict__ gb,
                                                    const float* __restrict__ mr,
                                                    u16b* __restrict__ xnT) {
    __shared__ __align__(16) u16b T[64 * 80];       // [n][c], stride 80 (160B rows)
    const int b = blockIdx.z, n0 = blockIdx.x * 64, c0 = blockIdx.y * 64;
    const float* xb = x + ((size_t)b * NC + c0) * NHW;
    const int tid = threadIdx.x;
    #pragma unroll
    for (int it = 0; it < 4; ++it) {                // read rows of x (coalesced f32x4)
        int tt = tid + it * 256;
        int cc = tt >> 4, nq = (tt & 15) * 4;
        float4 v = *(const float4*)(xb + (size_t)cc * NHW + n0 + nq);
        int c = c0 + cc;
        int g = (b * NC + c) >> 4;                  // 16 channels per group
        float m = mr[2 * g], rs = mr[2 * g + 1];
        float sc = gw[c] * rs, sb = gb[c] - m * sc;
        T[(nq + 0) * 80 + cc] = f2bf(v.x * sc + sb);
        T[(nq + 1) * 80 + cc] = f2bf(v.y * sc + sb);
        T[(nq + 2) * 80 + cc] = f2bf(v.z * sc + sb);
        T[(nq + 3) * 80 + cc] = f2bf(v.w * sc + sb);
    }
    __syncthreads();
    #pragma unroll
    for (int it = 0; it < 2; ++it) {                // write rows of xnT (b128)
        int tt = tid + it * 256;
        int nn = tt >> 3, co = (tt & 7) * 8;
        *(uint4*)(xnT + ((size_t)b * NHW + n0 + nn) * NC + c0 + co) =
            *(const uint4*)(T + nn * 80 + co);
    }
}

__global__ void __launch_bounds__(256) k_cvt_bf16(const float* __restrict__ src,
                                                  u16b* __restrict__ dst) {
    size_t idx = ((size_t)blockIdx.x * 256 + threadIdx.x) * 4;
    float4 v = *(const float4*)(src + idx);
    ushort4 o;
    o.x = f2bf(v.x); o.y = f2bf(v.y); o.z = f2bf(v.z); o.w = f2bf(v.w);
    *(ushort4*)(dst + idx) = o;
}

// ---------------------------------------------- projection GEMM (bf16, WMMA)
// D[m,n] = sum_k W[m,k]*XT[n,k] + bias[m]. Block tile 128x64, 8 waves.
// TRANS=1 -> out stored [n][m] (Q,K); TRANS=0 -> out stored [m][n] (V).
template <int TRANS>
__global__ void __launch_bounds__(256) k_gemm_proj(const u16b* __restrict__ Wb,
                                                   const float* __restrict__ bias,
                                                   const u16b* __restrict__ XT,
                                                   u16b* __restrict__ out) {
    __shared__ __align__(16) u16b As[128 * 32];     // [m][k]
    __shared__ __align__(16) u16b Bs[64 * 32];      // [n][k]
    const int n0 = blockIdx.x * 64, m0 = blockIdx.y * 128;
    const int tid = threadIdx.x, lane = tid & 31, w = tid >> 5;
    f32x8 acc[4] = {};
    for (int kk = 0; kk < NC; kk += 32) {
        #pragma unroll
        for (int it = 0; it < 2; ++it) {            // A: 128x32 = 512 b128 chunks
            int tt = tid + it * 256;
            int row = tt >> 2, cb = (tt & 3) * 8;
            async_copy16(As + row * 32 + cb, Wb + (size_t)(m0 + row) * NC + kk + cb);
        }
        {                                           // B: 64x32 = 256 b128 chunks
            int n = tid >> 2, cb = (tid & 3) * 8;
            async_copy16(Bs + n * 32 + cb, XT + (size_t)(n0 + n) * NC + kk + cb);
        }
        async_wait0();
        __syncthreads();
        bf16x16 a = load_frag(As + w * 16 * 32, 32, lane);
        #pragma unroll
        for (int nt = 0; nt < 4; ++nt) {
            bf16x16 bf = load_frag(Bs + nt * 16 * 32, 32, lane);
            acc[nt] = wmma_bf16(a, bf, acc[nt]);
        }
        __syncthreads();
    }
    const int h = lane >> 4, nl = lane & 15;
    const int mb = m0 + w * 16 + h * 8;
    #pragma unroll
    for (int nt = 0; nt < 4; ++nt) {
        int n = n0 + nt * 16 + nl;
        if (TRANS) {                                // [n][m]: pack pairs -> b32 stores
            u32b* po = (u32b*)(out + (size_t)n * NC + mb);
            #pragma unroll
            for (int rp = 0; rp < 4; ++rp) {
                u32b lo = f2bf(acc[nt][2 * rp] + bias[mb + 2 * rp]);
                u32b hi = f2bf(acc[nt][2 * rp + 1] + bias[mb + 2 * rp + 1]);
                po[rp] = lo | (hi << 16);
            }
        } else {                                    // [m][n]: lane-coalesced b16
            #pragma unroll
            for (int r = 0; r < 8; ++r)
                out[(size_t)(mb + r) * NHW + n] = f2bf(acc[nt][r] + bias[mb + r]);
        }
    }
}

// --------------------------------------- attention pass 1: row max & sum(exp)
__global__ void __launch_bounds__(256) k_attn_stats(const u16b* __restrict__ qT,
                                                    const u16b* __restrict__ kT,
                                                    float* __restrict__ ML) {
    __shared__ __align__(16) u16b Ks[64 * 512];     // K^T tile: [j][c]
    __shared__ float sM[8][64], sL[8][64];
    const int b = blockIdx.y, i0 = blockIdx.x * 64;
    const u16b* qb = qT + (size_t)b * NHW * NC;
    const u16b* kb = kT + (size_t)b * NHW * NC;
    const int tid = threadIdx.x, lane = tid & 31, w = tid >> 5;
    const int i_t = w & 3, nh = w >> 2;             // wave -> (i subtile, j half)
    const float scale = 0.044194173824159216f;      // 512^-0.5
    bf16x16 aF[16];                                 // loop-invariant Q^T fragments
    #pragma unroll
    for (int ks = 0; ks < 16; ++ks)
        aF[ks] = load_frag(qb + (size_t)(i0 + i_t * 16) * NC + ks * 32, NC, lane);
    float m_run[8], l_run[8];
    #pragma unroll
    for (int r = 0; r < 8; ++r) { m_run[r] = -3.0e38f; l_run[r] = 0.f; }

    for (int jt = 0; jt < 64; ++jt) {
        const int j0 = jt * 64;
        #pragma unroll
        for (int it = 0; it < 16; ++it) {           // async-stage K^T tile rows
            int tt = tid + it * 256;
            int jr = tt >> 6, co = (tt & 63) * 8;
            async_copy16(Ks + jr * 512 + co, kb + (size_t)(j0 + jr) * NC + co);
        }
        async_wait0();
        __syncthreads();
        f32x8 a0 = {}, a1 = {};
        #pragma unroll
        for (int ks = 0; ks < 16; ++ks) {
            bf16x16 b0 = load_frag(Ks + (nh * 32 + 0) * 512 + ks * 32, 512, lane);
            bf16x16 b1 = load_frag(Ks + (nh * 32 + 16) * 512 + ks * 32, 512, lane);
            a0 = wmma_bf16(aF[ks], b0, a0);
            a1 = wmma_bf16(aF[ks], b1, a1);
        }
        #pragma unroll
        for (int r = 0; r < 8; ++r) {               // online softmax stats
            float v0 = a0[r] * scale, v1 = a1[r] * scale;
            float t0 = v0, t1 = v1;
            #pragma unroll
            for (int msk = 1; msk < 16; msk <<= 1) {
                t0 = fmaxf(t0, __shfl_xor(t0, msk));
                t1 = fmaxf(t1, __shfl_xor(t1, msk));
            }
            float mnew = fmaxf(m_run[r], fmaxf(t0, t1));
            float s = __expf(v0 - mnew) + __expf(v1 - mnew);
            #pragma unroll
            for (int msk = 1; msk < 16; msk <<= 1) s += __shfl_xor(s, msk);
            l_run[r] = l_run[r] * __expf(m_run[r] - mnew) + s;
            m_run[r] = mnew;
        }
        __syncthreads();
    }
    if ((lane & 15) == 0) {
        int h = lane >> 4;
        #pragma unroll
        for (int r = 0; r < 8; ++r) {
            sM[w][i_t * 16 + h * 8 + r] = m_run[r];
            sL[w][i_t * 16 + h * 8 + r] = l_run[r];
        }
    }
    __syncthreads();
    if (tid < 64) {
        int it2 = tid >> 4;
        float ma = sM[it2][tid], mb2 = sM[it2 + 4][tid];
        float M = fmaxf(ma, mb2);
        float L = sL[it2][tid] * __expf(ma - M) + sL[it2 + 4][tid] * __expf(mb2 - M);
        float* dst = ML + ((size_t)b * NHW + i0 + tid) * 2;
        dst[0] = M; dst[1] = L;
    }
}

// -------------------- attention pass 2: O^T = softmax(Q^T K) V^T, flash style
// 512 threads / 16 waves; wave w -> i-subtile (w>>2) for BOTH phases,
// phase A col subtile (w&3), phase B 128-channel slice (w&3).
__global__ void __launch_bounds__(512) k_attn_out(const u16b* __restrict__ qT,
                                                  const u16b* __restrict__ kT,
                                                  const u16b* __restrict__ vmat,
                                                  const float* __restrict__ ML,
                                                  u16b* __restrict__ OT) {
    __shared__ __align__(16) u16b KV[64 * 512];     // K^T tile [j][c], then V tile [c][j]
    __shared__ __align__(16) u16b Ps[64 * 64];      // P tile bf16 [i][j]
    const int b = blockIdx.y, i0 = blockIdx.x * 64;
    const u16b* qb = qT + (size_t)b * NHW * NC;
    const u16b* kb = kT + (size_t)b * NHW * NC;
    const u16b* vb = vmat + (size_t)b * NC * NHW;
    const int tid = threadIdx.x, lane = tid & 31, w = tid >> 5;
    const int i_t = w >> 2, ntA = w & 3, cs = w & 3;
    const int h = lane >> 4, nl = lane & 15;
    const float scale = 0.044194173824159216f;

    float rowM[8], rowInvL[8];
    #pragma unroll
    for (int r = 0; r < 8; ++r) {
        const float* p = ML + ((size_t)b * NHW + i0 + i_t * 16 + h * 8 + r) * 2;
        rowM[r] = p[0];
        rowInvL[r] = 1.0f / p[1];
    }
    bf16x16 aF[16];                                 // loop-invariant Q^T fragments
    #pragma unroll
    for (int ks = 0; ks < 16; ++ks)
        aF[ks] = load_frag(qb + (size_t)(i0 + i_t * 16) * NC + ks * 32, NC, lane);

    f32x8 o[8] = {};
    for (int jt = 0; jt < 64; ++jt) {
        const int j0 = jt * 64;
        #pragma unroll
        for (int it = 0; it < 8; ++it) {            // async-stage K^T tile
            int tt = tid + it * 512;
            int jr = tt >> 6, co = (tt & 63) * 8;
            async_copy16(KV + jr * 512 + co, kb + (size_t)(j0 + jr) * NC + co);
        }
        async_wait0();
        __syncthreads();
        {                                           // phase A: S subtile -> exp -> Ps
            f32x8 s = {};
            #pragma unroll
            for (int ks = 0; ks < 16; ++ks) {
                bf16x16 bf = load_frag(KV + ntA * 16 * 512 + ks * 32, 512, lane);
                s = wmma_bf16(aF[ks], bf, s);
            }
            #pragma unroll
            for (int r = 0; r < 8; ++r) {
                float p = __expf(s[r] * scale - rowM[r]);
                Ps[(i_t * 16 + h * 8 + r) * 64 + ntA * 16 + nl] = f2bf(p);
            }
        }
        __syncthreads();
        #pragma unroll
        for (int it = 0; it < 8; ++it) {            // async-stage V tile (reuse KV)
            int tt = tid + it * 512;
            int c = tt >> 3, jb = (tt & 7) * 8;
            async_copy16(KV + c * 64 + jb, vb + (size_t)c * NHW + j0 + jb);
        }
        async_wait0();
        __syncthreads();
        #pragma unroll
        for (int kh = 0; kh < 2; ++kh) {            // phase B: O += P * V^T
            bf16x16 a = load_frag(Ps + i_t * 16 * 64 + kh * 32, 64, lane);
            #pragma unroll
            for (int nt = 0; nt < 8; ++nt) {
                bf16x16 bf = load_frag(KV + (cs * 128 + nt * 16) * 64 + kh * 32, 64, lane);
                o[nt] = wmma_bf16(a, bf, o[nt]);
            }
        }
        __syncthreads();
    }
    #pragma unroll
    for (int nt = 0; nt < 8; ++nt) {
        #pragma unroll
        for (int r = 0; r < 8; ++r) {
            int i = i0 + i_t * 16 + h * 8 + r;
            int c = cs * 128 + nt * 16 + nl;
            OT[((size_t)b * NHW + i) * NC + c] = f2bf(o[nt][r] * rowInvL[r]);
        }
    }
}

// ------------------------------------ output projection + bias + residual
__global__ void __launch_bounds__(256) k_proj_out(const u16b* __restrict__ Wb,
                                                  const float* __restrict__ bias,
                                                  const u16b* __restrict__ OT,
                                                  const float* __restrict__ x,
                                                  float* __restrict__ y) {
    __shared__ __align__(16) u16b As[128 * 32];
    __shared__ __align__(16) u16b Bs[64 * 32];
    const int b = blockIdx.z;
    const int n0 = blockIdx.x * 64, m0 = blockIdx.y * 128;
    const u16b* Ob = OT + (size_t)b * NHW * NC;
    const int tid = threadIdx.x, lane = tid & 31, w = tid >> 5;
    f32x8 acc[4] = {};
    for (int kk = 0; kk < NC; kk += 32) {
        #pragma unroll
        for (int it = 0; it < 2; ++it) {
            int tt = tid + it * 256;
            int row = tt >> 2, cb = (tt & 3) * 8;
            async_copy16(As + row * 32 + cb, Wb + (size_t)(m0 + row) * NC + kk + cb);
        }
        {
            int n = tid >> 2, cb = (tid & 3) * 8;
            async_copy16(Bs + n * 32 + cb, Ob + (size_t)(n0 + n) * NC + kk + cb);
        }
        async_wait0();
        __syncthreads();
        bf16x16 a = load_frag(As + w * 16 * 32, 32, lane);
        #pragma unroll
        for (int nt = 0; nt < 4; ++nt) {
            bf16x16 bf = load_frag(Bs + nt * 16 * 32, 32, lane);
            acc[nt] = wmma_bf16(a, bf, acc[nt]);
        }
        __syncthreads();
    }
    const int h = lane >> 4, nl = lane & 15;
    const float* xb = x + (size_t)b * NC * NHW;
    float* yb = y + (size_t)b * NC * NHW;
    const int mb = m0 + w * 16 + h * 8;
    #pragma unroll
    for (int nt = 0; nt < 4; ++nt) {
        int n = n0 + nt * 16 + nl;
        #pragma unroll
        for (int r = 0; r < 8; ++r)
            yb[(size_t)(mb + r) * NHW + n] = acc[nt][r] + bias[mb + r] + xb[(size_t)(mb + r) * NHW + n];
    }
}

// ---------------------------------------------------------------- launcher
extern "C" void kernel_launch(void* const* d_in, const int* in_sizes, int n_in,
                              void* d_out, int out_size, void* d_ws, size_t ws_size,
                              hipStream_t stream) {
    const float* x   = (const float*)d_in[0];
    const float* gnw = (const float*)d_in[1];
    const float* gnb = (const float*)d_in[2];
    const float* wq  = (const float*)d_in[3];
    const float* bq  = (const float*)d_in[4];
    const float* wk  = (const float*)d_in[5];
    const float* bk  = (const float*)d_in[6];
    const float* wv  = (const float*)d_in[7];
    const float* bv  = (const float*)d_in[8];
    const float* wp  = (const float*)d_in[9];
    const float* bp  = (const float*)d_in[10];
    float* y = (float*)d_out;

    char* ws = (char*)d_ws;
    size_t off = 0;
    float* mr = (float*)(ws + off); off += 1024;
    u16b* xnT = (u16b*)(ws + off); off += (size_t)NB * NHW * NC * 2;
    u16b* wqb = (u16b*)(ws + off); off += (size_t)NC * NC * 2;
    u16b* wkb = (u16b*)(ws + off); off += (size_t)NC * NC * 2;
    u16b* wvb = (u16b*)(ws + off); off += (size_t)NC * NC * 2;
    u16b* wpb = (u16b*)(ws + off); off += (size_t)NC * NC * 2;
    u16b* qT  = (u16b*)(ws + off); off += (size_t)NB * NHW * NC * 2;
    u16b* kT  = (u16b*)(ws + off); off += (size_t)NB * NHW * NC * 2;
    u16b* vM  = (u16b*)(ws + off); off += (size_t)NB * NC * NHW * 2;
    float* ML = (float*)(ws + off); off += (size_t)NB * NHW * 2 * 4;
    u16b* OT  = (u16b*)(ws + off); off += (size_t)NB * NHW * NC * 2;

    k_gn_stats<<<NB * NG, 256, 0, stream>>>(x, mr);
    k_gn_apply_t<<<dim3(NHW / 64, NC / 64, NB), 256, 0, stream>>>(x, gnw, gnb, mr, xnT);
    k_cvt_bf16<<<(NC * NC) / 1024, 256, 0, stream>>>(wq, wqb);
    k_cvt_bf16<<<(NC * NC) / 1024, 256, 0, stream>>>(wk, wkb);
    k_cvt_bf16<<<(NC * NC) / 1024, 256, 0, stream>>>(wv, wvb);
    k_cvt_bf16<<<(NC * NC) / 1024, 256, 0, stream>>>(wp, wpb);

    for (int b = 0; b < NB; ++b) {
        const u16b* xb = xnT + (size_t)b * NHW * NC;
        k_gemm_proj<1><<<dim3(NHW / 64, NC / 128), 256, 0, stream>>>(wqb, bq, xb, qT + (size_t)b * NHW * NC);
        k_gemm_proj<1><<<dim3(NHW / 64, NC / 128), 256, 0, stream>>>(wkb, bk, xb, kT + (size_t)b * NHW * NC);
        k_gemm_proj<0><<<dim3(NHW / 64, NC / 128), 256, 0, stream>>>(wvb, bv, xb, vM + (size_t)b * NC * NHW);
    }

    k_attn_stats<<<dim3(NHW / 64, NB), 256, 0, stream>>>(qT, kT, ML);
    k_attn_out<<<dim3(NHW / 64, NB), 512, 0, stream>>>(qT, kT, vM, ML, OT);
    k_proj_out<<<dim3(NHW / 64, NC / 128, NB), 256, 0, stream>>>(wpb, bp, OT, x, y);
}
